// AttnBlock_71579924955751
// MI455X (gfx1250) — matrix operand order repaired and checked by reading, not compiled, not run on hardware
//
#include <hip/hip_runtime.h>
#include <hip/hip_bf16.h>

typedef __attribute__((ext_vector_type(16))) _Float16 v16h;
typedef __attribute__((ext_vector_type(8)))  _Float16 v8h;
typedef __attribute__((ext_vector_type(8)))  float    v8f;

#define SEQ_L 4096
#define NHEAD 8
#define HDIM  16

static __device__ __forceinline__ v8f wmma_f16(v16h a, v16h b, v8f c) {
  return __builtin_amdgcn_wmma_f32_16x16x32_f16(false, a, false, b, (short)0, c, false, false);
}

// ------------------------------------------------------------------
// Kernel 1: qkv = x^T @ w_qkv^T + b_qkv, packed into q,k (row-major
// [head][l][d], f16) and v transposed ([head][d][l], f16).
// Channel split is the reference's d-major order: j = dd*24 + ff*3 + kk.
// The softmax scale sqrt(d)=4 is folded into the stored Q (exact).
// ------------------------------------------------------------------
__global__ __launch_bounds__(256) void qkv_proj_kernel(
    const float* __restrict__ x, const float* __restrict__ w_qkv,
    const float* __restrict__ b_qkv,
    _Float16* __restrict__ q, _Float16* __restrict__ k,
    _Float16* __restrict__ vt)
{
  const int j = blockIdx.y;                      // 0..383 (uniform -> scalar weight loads)
  const int l = blockIdx.x * 256 + threadIdx.x;  // 0..4095
  const float* __restrict__ wrow = w_qkv + j * 128;
  float acc = b_qkv[j];
  #pragma unroll 16
  for (int c = 0; c < 128; ++c)
    acc = fmaf(x[(size_t)c * SEQ_L + l], wrow[c], acc);
  const int kk = j % 3;
  const int ff = (j / 3) % 8;
  const int dd = j / 24;
  if (kk == 0)      q[((size_t)ff * SEQ_L + l) * HDIM + dd] = (_Float16)(acc * 4.0f);
  else if (kk == 1) k[((size_t)ff * SEQ_L + l) * HDIM + dd] = (_Float16)acc;
  else              vt[((size_t)(ff * HDIM + dd)) * SEQ_L + l] = (_Float16)acc;
}

// ------------------------------------------------------------------
// Kernel 2: flash attention, one wave per 16-row query tile.
// Transposed formulation: S^T = K Q^T, O^T = V^T P^T, so each lane owns
// one query column -> softmax is (almost) lane-local.
// ------------------------------------------------------------------
__global__ __launch_bounds__(128) void flash_attn_kernel(
    const _Float16* __restrict__ q4, const _Float16* __restrict__ k,
    const _Float16* __restrict__ vt, float* __restrict__ obuf)
{
  // per-wave P^T staging: [query 0..15][key 0..31], stride 40 halves (80B)
  __shared__ __align__(16) _Float16 lds[4][16 * 40];

  const int wave  = threadIdx.x >> 5;
  const int lane  = threadIdx.x & 31;
  const int sub   = lane >> 4;   // half-wave select
  const int l16   = lane & 15;
  const int head  = blockIdx.y;
  const int qbase = (blockIdx.x * 4 + wave) * 16;

  const _Float16* __restrict__ qh = q4 + (size_t)head * SEQ_L * HDIM;
  const _Float16* __restrict__ kh = k  + (size_t)head * SEQ_L * HDIM;
  const _Float16* __restrict__ vh = vt + (size_t)head * HDIM * SEQ_L;
  _Float16* plds = &lds[wave][0];

  // --- Q^T as B-fragment (32x16): lane col = query l16; kk=0..15 = d
  //     (h0..7 = d[sub*8..]), kk=16..31 is the zero padding (h8..15) ---
  v16h b_q;
  {
    const v8h* qp = (const v8h*)(qh + (size_t)(qbase + l16) * HDIM);
    v8h q0 = qp[0], q1 = qp[1];
    #pragma unroll
    for (int i = 0; i < 8; ++i) {
      b_q[i]     = sub ? q1[i] : q0[i];
      b_q[8 + i] = (_Float16)0.f;
    }
  }

  float m_run = -3.0e38f;   // running max (synced across lane pair)
  float l_run = 0.f;        // running partial sum (this lane's keys only)
  v8f o_acc = {};           // O^T fragment: rows=dd, cols=queries

  for (int jbase = 0; jbase < SEQ_L; jbase += 32) {
    // --- K tiles as A-fragments (16x32): lane row = key; h0..7 = d[sub*8..] ---
    v16h a_k0, a_k1;
    {
      v8h kv0 = *(const v8h*)(kh + (size_t)(jbase + l16) * HDIM + sub * 8);
      v8h kv1 = *(const v8h*)(kh + (size_t)(jbase + 16 + l16) * HDIM + sub * 8);
      #pragma unroll
      for (int i = 0; i < 8; ++i) {
        a_k0[i] = kv0[i]; a_k0[8 + i] = (_Float16)0.f;
        a_k1[i] = kv1[i]; a_k1[8 + i] = (_Float16)0.f;
      }
    }
    v8f c0 = {}, c1 = {};
    c0 = wmma_f16(a_k0, b_q, c0);   // S^T tile0: elem r -> key r+sub*8, col=query
    c1 = wmma_f16(a_k1, b_q, c1);   // S^T tile1 (keys +16)

    // --- lane-local online softmax for query l16 over this lane's 16 keys ---
    float mx = fmaxf(c0[0], c0[1]);
    #pragma unroll
    for (int i = 2; i < 8; ++i) mx = fmaxf(mx, c0[i]);
    #pragma unroll
    for (int i = 0; i < 8; ++i) mx = fmaxf(mx, c1[i]);
    mx = fmaxf(mx, __shfl_xor(mx, 16, 32));        // combine with partner lane
    const float mnew  = fmaxf(m_run, mx);
    const float alpha = __expf(m_run - mnew);
    m_run = mnew;

    float rs = 0.f;
    v8h ph0, ph1;
    #pragma unroll
    for (int i = 0; i < 8; ++i) {
      float p = __expf(c0[i] - mnew); rs += p; ph0[i] = (_Float16)p;
    }
    #pragma unroll
    for (int i = 0; i < 8; ++i) {
      float p = __expf(c1[i] - mnew); rs += p; ph1[i] = (_Float16)p;
    }
    l_run = l_run * alpha + rs;
    #pragma unroll
    for (int r = 0; r < 8; ++r) o_acc[r] *= alpha;

    // --- stage P^T: row-major [query][key]; this lane owns contiguous 16B runs
    *(v8h*)(plds + l16 * 40 + sub * 8)      = ph0;   // tile0 keys sub*8..+7
    *(v8h*)(plds + l16 * 40 + 16 + sub * 8) = ph1;   // tile1 keys sub*8..+7
    asm volatile("s_wait_dscnt 0" ::: "memory");

    // --- P^T as B-fragment: lane col = query l16, kk = keys (sub*8.., 16+sub*8..) ---
    v16h b_p;
    {
      const v8h* pp = (const v8h*)(plds + l16 * 40 + sub * 8);
      v8h p0 = pp[0], p1 = pp[2];
      #pragma unroll
      for (int i = 0; i < 8; ++i) { b_p[i] = p0[i]; b_p[8 + i] = p1[i]; }
    }
    // --- V^T as A-fragment (16x32): lane row = dd, K = keys ---
    v16h a_v;
    {
      const _Float16* vrow = vh + (size_t)l16 * SEQ_L + jbase;
      v8h v0 = *(const v8h*)(vrow + sub * 8);
      v8h v1 = *(const v8h*)(vrow + 16 + sub * 8);
      #pragma unroll
      for (int i = 0; i < 8; ++i) { a_v[i] = v0[i]; a_v[8 + i] = v1[i]; }
    }
    o_acc = wmma_f16(a_v, b_p, o_acc);   // O^T += V^T P^T
  }

  // --- normalize (lane-local) and write O as [channel = head*16+dd][l] ---
  const float l_tot = l_run + __shfl_xor(l_run, 16, 32);
  const float inv = 1.0f / l_tot;
  #pragma unroll
  for (int r = 0; r < 8; ++r) {
    obuf[(size_t)(head * HDIM + r + sub * 8) * SEQ_L + qbase + l16] = o_acc[r] * inv;
  }
}

// ------------------------------------------------------------------
// Kernel 3: out[c,h,:] = o[c,h,:] @ w_proj^T + b_proj  (over W axis,
// faithful to the reference). One block per row, o-row is wave-uniform.
// ------------------------------------------------------------------
__global__ __launch_bounds__(128) void proj_kernel(
    const float* __restrict__ obuf, const float* __restrict__ w_proj,
    const float* __restrict__ b_proj, float* __restrict__ out)
{
  const int row = blockIdx.x;      // (c*32 + h), 0..4095
  const int j   = threadIdx.x;     // output w index 0..127
  const float* __restrict__ orow = obuf + (size_t)row * 128;
  const float* __restrict__ wrow = w_proj + j * 128;
  float acc = b_proj[j];
  #pragma unroll 16
  for (int w = 0; w < 128; ++w)
    acc = fmaf(orow[w], wrow[w], acc);
  out[(size_t)row * 128 + j] = acc;
}

extern "C" void kernel_launch(void* const* d_in, const int* in_sizes, int n_in,
                              void* d_out, int out_size, void* d_ws, size_t ws_size,
                              hipStream_t stream) {
  const float* x      = (const float*)d_in[0];
  const float* w_qkv  = (const float*)d_in[1];
  const float* b_qkv  = (const float*)d_in[2];
  const float* w_proj = (const float*)d_in[3];
  const float* b_proj = (const float*)d_in[4];
  float* out = (float*)d_out;

  char* ws = (char*)d_ws;
  _Float16* q  = (_Float16*)(ws);                       // 1 MB (pre-scaled by 4)
  _Float16* k  = (_Float16*)(ws + (1u << 20));          // 1 MB
  _Float16* vt = (_Float16*)(ws + (2u << 20));          // 1 MB
  float*    ob = (float*)   (ws + (3u << 20));          // 2 MB

  // 1) QKV projection + pack (grid: 16 l-tiles x 384 channels)
  qkv_proj_kernel<<<dim3(SEQ_L / 256, 384), 256, 0, stream>>>(x, w_qkv, b_qkv, q, k, vt);

  // 2) flash attention: 256 q-tiles (4 waves/block) x 8 heads
  flash_attn_kernel<<<dim3(SEQ_L / 16 / 4, NHEAD), 128, 0, stream>>>(q, k, vt, ob);

  // 3) W-axis projection
  proj_kernel<<<dim3(128 * 32), 128, 0, stream>>>(ob, w_proj, b_proj, out);
}
